// BinaryLinear_8693013807478
// MI455X (gfx1250) — matrix operand order repaired and checked by reading, not compiled, not run on hardware
//
#include <hip/hip_runtime.h>
#include <hip/hip_bf16.h>
#include <stdint.h>

typedef __attribute__((ext_vector_type(16))) _Float16 v16h;
typedef __attribute__((ext_vector_type(8)))  _Float16 v8h;
typedef __attribute__((ext_vector_type(8)))  float    v8f;
typedef __attribute__((ext_vector_type(4)))  float    v4f;
typedef __attribute__((ext_vector_type(4)))  int      v4i;

#define AS1 __attribute__((address_space(1)))
#define AS3 __attribute__((address_space(3)))

#define MDIM 8192
#define KDIM 2048
#define NDIM 2048

#define BM 128
#define BN 128
#define BK 32
#define LDROW 80            // LDS row stride in bytes (64 data + 16 pad -> conflict-free)
#define ABUF  (128 * LDROW) // one A (or B) tile buffer: 10240 B

#if __has_builtin(__builtin_amdgcn_global_load_async_to_lds_b128)
#define HAVE_ASYNC 1
#else
#define HAVE_ASYNC 0
#endif

__device__ __forceinline__ void copy16_g2l(const _Float16* g, char* l) {
#if HAVE_ASYNC
    // param types (from hipcc diagnostic): (AS1 v4i*, AS3 v4i*, imm offset, imm cpol)
    // int->pointer casts: flat->AS1 value-identical; flat->AS3 truncates to LDS offset.
    __builtin_amdgcn_global_load_async_to_lds_b128(
        (AS1 v4i*)(uintptr_t)g, (AS3 v4i*)(uintptr_t)l, 0, 0);
#else
    *(v8h*)l = *(const v8h*)g;   // sync fallback: global->VGPR->LDS
#endif
}

template <int N>
__device__ __forceinline__ void wait_async() {
#if HAVE_ASYNC
#if __has_builtin(__builtin_amdgcn_s_wait_asynccnt)
    __builtin_amdgcn_s_wait_asynccnt(N);
#else
    asm volatile("s_wait_asynccnt %0" ::"i"(N) : "memory");
#endif
#endif
}

// ---------------- pass 1a: x f32 -> f16 ----------------
__global__ __launch_bounds__(256) void cvt_x_kernel(const float* __restrict__ x,
                                                    _Float16* __restrict__ xh) {
    size_t i = ((size_t)blockIdx.x * 256 + threadIdx.x) * 8;
    v4f a = *(const v4f*)(x + i);
    v4f b = *(const v4f*)(x + i + 4);
    v8h o;
    o[0] = (_Float16)a[0]; o[1] = (_Float16)a[1];
    o[2] = (_Float16)a[2]; o[3] = (_Float16)a[3];
    o[4] = (_Float16)b[0]; o[5] = (_Float16)b[1];
    o[6] = (_Float16)b[2]; o[7] = (_Float16)b[3];
    *(v8h*)(xh + i) = o;
}

// ------- pass 1b: W f32 -> binarized f16, transposed (Wt[n][k]) -------
__global__ __launch_bounds__(256) void binw_kernel(const float* __restrict__ w,
                                                   _Float16* __restrict__ wt) {
    __shared__ _Float16 t[32][40];       // padded rows: no bank conflicts
    const int tx = threadIdx.x & 31;
    const int ty = threadIdx.x >> 5;     // 0..7
    const int n0 = blockIdx.x * 32;
    const int k0 = blockIdx.y * 32;
#pragma unroll
    for (int i = 0; i < 4; ++i) {
        int r = ty + i * 8;
        float v = w[(size_t)(k0 + r) * NDIM + n0 + tx];
        t[r][tx] = (v > 0.0f) ? (_Float16)1 : (_Float16)0;
    }
    __syncthreads();
#pragma unroll
    for (int i = 0; i < 4; ++i) {
        int r = ty + i * 8;
        wt[(size_t)(n0 + r) * KDIM + k0 + tx] = t[tx][r];
    }
}

// ---------------- pass 2: WMMA GEMM  out = Xh @ Wt^T ----------------
__global__ __launch_bounds__(256) void gemm_kernel(const _Float16* __restrict__ Xh,
                                                   const _Float16* __restrict__ Wt,
                                                   float* __restrict__ out) {
    __shared__ __align__(16) char smem[4 * ABUF];  // A0 A1 B0 B1

    const int tid  = threadIdx.x;
    const int lane = tid & 31;
    const int wave = tid >> 5;
    const int wm   = wave & 3;   // wave row-slab: 4 x 32 rows
    const int wn   = wave >> 2;  // wave col-slab: 2 x 64 cols
    const int g    = lane >> 4;  // half-wave group
    const int m16  = lane & 15;

    const int blockN = blockIdx.x * BN;
    const int blockM = blockIdx.y * BM;

    // issue copies for one K-tile into buffer `buf` (4 async b128 per thread)
    auto issue = [&](int buf, int k0) {
        char* lA = smem + buf * ABUF;
        char* lB = smem + (2 + buf) * ABUF;
#pragma unroll
        for (int i = 0; i < 2; ++i) {
            int c = tid + i * 256;       // chunk id, 512 chunks of 16B per tile
            int row = c >> 2;
            int off = c & 3;             // 16B unit within 64B row
            copy16_g2l(Xh + (size_t)(blockM + row) * KDIM + k0 + off * 8,
                       lA + row * LDROW + off * 16);
            copy16_g2l(Wt + (size_t)(blockN + row) * KDIM + k0 + off * 8,
                       lB + row * LDROW + off * 16);
        }
    };

    v8f acc[2][4] = {};

    issue(0, 0);
    for (int kt = 0; kt < KDIM / BK; ++kt) {
        const int buf = kt & 1;
        if (kt + 1 < KDIM / BK) {
            issue(buf ^ 1, (kt + 1) * BK);
            wait_async<4>();             // current tile's 4 ops done; next 4 in flight
        } else {
            wait_async<0>();
        }
        __syncthreads();

        const char* lA = smem + buf * ABUF;
        const char* lB = smem + (2 + buf) * ABUF;

        v16h af[2], bf[4];
#pragma unroll
        for (int sm = 0; sm < 2; ++sm) {
            // A 16x32 f16 layout: lane (g,m) holds K[g*8..g*8+8) then K[16+g*8 ..)
            const char* ar = lA + (wm * 32 + sm * 16 + m16) * LDROW + g * 16;
            v8h lo = *(const v8h*)ar;
            v8h hi = *(const v8h*)(ar + 32);
            af[sm] = __builtin_shufflevector(lo, hi, 0, 1, 2, 3, 4, 5, 6, 7, 8, 9,
                                             10, 11, 12, 13, 14, 15);
        }
#pragma unroll
        for (int sn = 0; sn < 4; ++sn) {
            // B 32x16 f16 layout: lane (g,n) holds column n, K[g*8..g*8+8) then K[16+g*8 ..)
            const char* br = lB + (wn * 64 + sn * 16 + m16) * LDROW + g * 16;
            v8h lo = *(const v8h*)br;
            v8h hi = *(const v8h*)(br + 32);
            bf[sn] = __builtin_shufflevector(lo, hi, 0, 1, 2, 3, 4, 5, 6, 7, 8, 9,
                                             10, 11, 12, 13, 14, 15);
        }
#pragma unroll
        for (int sm = 0; sm < 2; ++sm)
#pragma unroll
            for (int sn = 0; sn < 4; ++sn)
                acc[sm][sn] = __builtin_amdgcn_wmma_f32_16x16x32_f16(
                    false, af[sm], false, bf[sn], (short)0, acc[sm][sn], false, false);
        __syncthreads();
    }

    // epilogue: C/D layout -> lane(g,n), VGPR v: row = base + 8g + v, col = base + n
#pragma unroll
    for (int sm = 0; sm < 2; ++sm) {
        const int row0 = blockM + wm * 32 + sm * 16 + g * 8;
#pragma unroll
        for (int sn = 0; sn < 4; ++sn) {
            const int col = blockN + wn * 64 + sn * 16 + m16;
#pragma unroll
            for (int v = 0; v < 8; ++v)
                out[(size_t)(row0 + v) * NDIM + col] = acc[sm][sn][v];
        }
    }
}

extern "C" void kernel_launch(void* const* d_in, const int* in_sizes, int n_in,
                              void* d_out, int out_size, void* d_ws, size_t ws_size,
                              hipStream_t stream) {
    const float* x = (const float*)d_in[0];
    const float* w = (const float*)d_in[1];
    float* out = (float*)d_out;

    char* ws = (char*)d_ws;
    _Float16* Xh = (_Float16*)ws;                                   // 32 MB
    _Float16* Wt = (_Float16*)(ws + (size_t)MDIM * KDIM * 2);       //  8 MB

    cvt_x_kernel<<<(MDIM * KDIM) / (256 * 8), 256, 0, stream>>>(x, Xh);
    binw_kernel<<<dim3(NDIM / 32, KDIM / 32), 256, 0, stream>>>(w, Wt);
    gemm_kernel<<<dim3(NDIM / BN, MDIM / BM), 256, 0, stream>>>(Xh, Wt, out);
}
